// RNN_13245679141113
// MI455X (gfx1250) — matrix operand order repaired and checked by reading, not compiled, not run on hardware
//
#include <hip/hip_runtime.h>

typedef __attribute__((ext_vector_type(16))) __bf16 v16bf;
typedef __attribute__((ext_vector_type(8)))  __bf16 v8bf;
typedef __attribute__((ext_vector_type(4)))  __bf16 v4bf;
typedef __attribute__((ext_vector_type(8)))  float  v8f;

#define NVOC   32000
#define NINPUT 512
#define NHID   1024
#define BB     16
#define TT     256
#define NG     (4 * NHID)      // 4096 gate width
#define MTOT   (BB * TT)       // 4096 rows (b*T + t)

// ---------------------------------------------------------------------------
// WMMA fragment loaders (wave32, V_WMMA_F32_16X16X32_BF16 layouts per ISA 7.12.2)
// ---------------------------------------------------------------------------
static __device__ __forceinline__ v16bf cat16(v8bf lo, v8bf hi) {
  return __builtin_shufflevector(lo, hi, 0,1,2,3,4,5,6,7,8,9,10,11,12,13,14,15);
}

// A: 16x32 bf16 tile from row-major matrix (ld = row stride in elements).
// lane: m = lane&15, hi = lane>>4 ; elems[0:8) K=hi*8+e ; elems[8:16) K=16+hi*8+e'
static __device__ __forceinline__ v16bf load_a_frag(const __bf16* __restrict__ base,
                                                    int ld, int lane) {
  int m = lane & 15, hi = lane >> 4;
  const __bf16* p = base + (size_t)m * ld + hi * 8;
  v8bf lo  = *(const v8bf*)(p);
  v8bf hi8 = *(const v8bf*)(p + 16);
  return cat16(lo, hi8);
}

// B: 32(K)x16(N) tile from N-major storage (row n holds K contiguous).
// lane: n = lane&15, hi = lane>>4 ; elems[0:8) K=hi*8+e ; elems[8:16) K=16+hi*8+e'
static __device__ __forceinline__ v16bf load_b_frag(const __bf16* __restrict__ base,
                                                    int ld, int lane) {
  int n = lane & 15, hi = lane >> 4;
  const __bf16* p = base + (size_t)n * ld + hi * 8;
  v8bf lo  = *(const v8bf*)(p);
  v8bf hi8 = *(const v8bf*)(p + 16);
  return cat16(lo, hi8);
}

// ---------------------------------------------------------------------------
// Prep kernels: embedding gather + bf16 conversions / transposes
// ---------------------------------------------------------------------------
__global__ void embed_gather(const int* __restrict__ tok, const float* __restrict__ E,
                             __bf16* __restrict__ out) {
  int bt = blockIdx.x;                       // row = b*T + t
  int row = tok[bt];
  const float* src = E + (size_t)row * NINPUT;
  __bf16* dst = out + (size_t)bt * NINPUT;
  for (int d = threadIdx.x; d < NINPUT; d += blockDim.x) dst[d] = (__bf16)src[d];
}

// fp32 (R x C) row-major -> bf16 (C x R) row-major, LDS-tiled transpose
__global__ void transpose_cvt(const float* __restrict__ in, __bf16* __restrict__ out,
                              int R, int C) {
  __shared__ float tile[32][33];
  int c0 = blockIdx.x * 32, r0 = blockIdx.y * 32;
  for (int i = threadIdx.y; i < 32; i += 8)
    tile[i][threadIdx.x] = in[(size_t)(r0 + i) * C + c0 + threadIdx.x];
  __syncthreads();
  for (int i = threadIdx.y; i < 32; i += 8)
    out[(size_t)(c0 + i) * R + r0 + threadIdx.x] = (__bf16)tile[threadIdx.x][i];
}

// straight fp32 -> bf16 (dec_W already N-major: (NVOC, NHID))
__global__ void cvt_bf16(const float* __restrict__ in, __bf16* __restrict__ out,
                         size_t n4) {
  size_t i = (size_t)blockIdx.x * blockDim.x + threadIdx.x;
  if (i >= n4) return;
  float4 v = ((const float4*)in)[i];
  v4bf o; o[0] = (__bf16)v.x; o[1] = (__bf16)v.y; o[2] = (__bf16)v.z; o[3] = (__bf16)v.w;
  *(v4bf*)(out + 4 * i) = o;
}

__global__ void zero_state(float* __restrict__ c, __bf16* __restrict__ hb,
                           float* __restrict__ hf) {
  int i = blockIdx.x * blockDim.x + threadIdx.x;   // 16384 elems
  c[i] = 0.f; hf[i] = 0.f; hb[i] = (__bf16)0.f;
}

// ---------------------------------------------------------------------------
// Big GEMM: C(MxN,f32) = A(MxK,bf16 row-major) * B(NxK,bf16 N-major)^T + bias[N]
// block = 8 waves covering 128(M) x 128(N); each wave: 64(M) x 32(N) (4x2 tiles).
// Tall wave tile minimizes B traffic (B = dec_W, the 64 MB L2-resident stream);
// extra A traffic hits the small hot h_all set (8 MB) in L1/L2.
// Requires M%128==0, N%128==0, K%32==0.
// ---------------------------------------------------------------------------
__global__ __launch_bounds__(256) void gemm_bf16_nt(
    const __bf16* __restrict__ A, const __bf16* __restrict__ Bm,
    const float* __restrict__ bias, float* __restrict__ C,
    int K, int lda, int ldb, int ldc) {
  int lane = threadIdx.x & 31;
  int wave = threadIdx.x >> 5;                       // 0..7
  int m0 = blockIdx.y * 128 + (wave & 1) * 64;
  int n0 = blockIdx.x * 128 + (wave >> 1) * 32;

  v8f acc[4][2];
#pragma unroll
  for (int i = 0; i < 4; ++i)
#pragma unroll
    for (int j = 0; j < 2; ++j) acc[i][j] = (v8f){0,0,0,0,0,0,0,0};

#pragma unroll 4
  for (int kk = 0; kk < K; kk += 32) {
    v16bf a0 = load_a_frag(A + (size_t)(m0 +  0) * lda + kk, lda, lane);
    v16bf a1 = load_a_frag(A + (size_t)(m0 + 16) * lda + kk, lda, lane);
    v16bf a2 = load_a_frag(A + (size_t)(m0 + 32) * lda + kk, lda, lane);
    v16bf a3 = load_a_frag(A + (size_t)(m0 + 48) * lda + kk, lda, lane);
    v16bf b0 = load_b_frag(Bm + (size_t)(n0 +  0) * ldb + kk, ldb, lane);
    v16bf b1 = load_b_frag(Bm + (size_t)(n0 + 16) * ldb + kk, ldb, lane);
    acc[0][0] = __builtin_amdgcn_wmma_f32_16x16x32_bf16(false, a0, false, b0, (short)0, acc[0][0], false, false);
    acc[0][1] = __builtin_amdgcn_wmma_f32_16x16x32_bf16(false, a0, false, b1, (short)0, acc[0][1], false, false);
    acc[1][0] = __builtin_amdgcn_wmma_f32_16x16x32_bf16(false, a1, false, b0, (short)0, acc[1][0], false, false);
    acc[1][1] = __builtin_amdgcn_wmma_f32_16x16x32_bf16(false, a1, false, b1, (short)0, acc[1][1], false, false);
    acc[2][0] = __builtin_amdgcn_wmma_f32_16x16x32_bf16(false, a2, false, b0, (short)0, acc[2][0], false, false);
    acc[2][1] = __builtin_amdgcn_wmma_f32_16x16x32_bf16(false, a2, false, b1, (short)0, acc[2][1], false, false);
    acc[3][0] = __builtin_amdgcn_wmma_f32_16x16x32_bf16(false, a3, false, b0, (short)0, acc[3][0], false, false);
    acc[3][1] = __builtin_amdgcn_wmma_f32_16x16x32_bf16(false, a3, false, b1, (short)0, acc[3][1], false, false);
  }

  int cn = lane & 15, chi = lane >> 4;               // C layout: row = v + chi*8
#pragma unroll
  for (int j = 0; j < 2; ++j) {
    float bj = bias[n0 + j * 16 + cn];
#pragma unroll
    for (int i = 0; i < 4; ++i)
#pragma unroll
      for (int v = 0; v < 8; ++v)
        C[(size_t)(m0 + i * 16 + chi * 8 + v) * ldc + n0 + j * 16 + cn] =
            acc[i][j][v] + bj;
  }
}

// ---------------------------------------------------------------------------
// LSTM recurrence step, GEMM part: gates(16x4096) = xg[:,t,:] + h(16x1024)@W_hh
// 256 n-tiles of 16x16; K = 1024 -> 32 chained WMMAs, C seeded from xg.
// grid = 32 blocks x 8 waves.
// ---------------------------------------------------------------------------
__global__ __launch_bounds__(256) void lstm_gemm_hh(
    const __bf16* __restrict__ h, const __bf16* __restrict__ Whh,
    const float* __restrict__ xg, float* __restrict__ gates, int t) {
  int lane = threadIdx.x & 31;
  int wave = threadIdx.x >> 5;
  int n0 = (blockIdx.x * 8 + wave) * 16;
  int cn = lane & 15, chi = lane >> 4;

  v8f acc;
#pragma unroll
  for (int v = 0; v < 8; ++v) {
    int m = chi * 8 + v;                              // batch index
    acc[v] = xg[((size_t)m * TT + t) * NG + n0 + cn];
  }
#pragma unroll 4
  for (int kk = 0; kk < NHID; kk += 32) {
    v16bf a = load_a_frag(h + kk, NHID, lane);
    v16bf b = load_b_frag(Whh + (size_t)n0 * NHID + kk, NHID, lane);
    acc = __builtin_amdgcn_wmma_f32_16x16x32_bf16(false, a, false, b, (short)0, acc, false, false);
  }
#pragma unroll
  for (int v = 0; v < 8; ++v)
    gates[(size_t)(chi * 8 + v) * NG + n0 + cn] = acc[v];
}

// LSTM pointwise: c,h update; h emitted in bf16 for next GEMM + decoder input
__global__ void lstm_pointwise(const float* __restrict__ gates, float* __restrict__ c,
                               float* __restrict__ hf, __bf16* __restrict__ hb,
                               __bf16* __restrict__ hall, int t) {
  int idx = blockIdx.x * blockDim.x + threadIdx.x;    // 0..16383
  int b = idx >> 10, j = idx & 1023;
  const float* g = gates + (size_t)b * NG;
  float gi = g[j], gf = g[j + NHID], gg = g[j + 2 * NHID], go = g[j + 3 * NHID];
  float si = 1.f / (1.f + __expf(-gi));
  float sf = 1.f / (1.f + __expf(-gf));
  float tg = tanhf(gg);
  float so = 1.f / (1.f + __expf(-go));
  float cnew = sf * c[idx] + si * tg;
  float hnew = so * tanhf(cnew);
  c[idx] = cnew;
  hf[idx] = hnew;
  __bf16 hv = (__bf16)hnew;
  hb[idx] = hv;
  hall[((size_t)b * TT + t) * NHID + j] = hv;
}

__global__ void copy_hidden(const float* __restrict__ hf, const float* __restrict__ cf,
                            float* __restrict__ out) {
  int i = blockIdx.x * blockDim.x + threadIdx.x;      // 16384
  out[i] = hf[i];
  out[BB * NHID + i] = cf[i];
}

// ---------------------------------------------------------------------------
extern "C" void kernel_launch(void* const* d_in, const int* in_sizes, int n_in,
                              void* d_out, int out_size, void* d_ws, size_t ws_size,
                              hipStream_t stream) {
  const int*   tokens = (const int*)d_in[0];
  const float* embW   = (const float*)d_in[1];
  const float* W_ih   = (const float*)d_in[2];
  const float* W_hh   = (const float*)d_in[3];
  const float* bias   = (const float*)d_in[4];
  const float* decW   = (const float*)d_in[5];
  const float* decb   = (const float*)d_in[6];
  float* out = (float*)d_out;

  char* ws = (char*)d_ws;
  size_t off = 0;
  auto salloc = [&](size_t bytes) -> void* {
    void* p = ws + off;
    off = (off + bytes + 255) & ~(size_t)255;
    return p;
  };
  __bf16* emb_b  = (__bf16*)salloc((size_t)MTOT * NINPUT * 2);   // 4 MB  (B*T,512)
  __bf16* Wih_t  = (__bf16*)salloc((size_t)NG * NINPUT * 2);     // 4 MB  (4096,512) N-major
  __bf16* Whh_t  = (__bf16*)salloc((size_t)NG * NHID * 2);       // 8 MB  (4096,1024) N-major
  __bf16* decW_b = (__bf16*)salloc((size_t)NVOC * NHID * 2);     // 64 MB (32000,1024) N-major
  float*  xg     = (float*)salloc((size_t)MTOT * NG * 4);        // 64 MB (B*T,4096)
  float*  gates  = (float*)salloc((size_t)BB * NG * 4);          // 256 KB
  __bf16* h_b    = (__bf16*)salloc((size_t)BB * NHID * 2);       // current h, bf16
  __bf16* hall_b = (__bf16*)salloc((size_t)MTOT * NHID * 2);     // 8 MB all h
  float*  h_f    = (float*)salloc((size_t)BB * NHID * 4);
  float*  c_f    = (float*)salloc((size_t)BB * NHID * 4);

  // h0 = c0 = 0
  zero_state<<<64, 256, 0, stream>>>(c_f, h_b, h_f);
  // embedding gather -> bf16
  embed_gather<<<MTOT, 128, 0, stream>>>(tokens, embW, emb_b);
  // weights -> bf16, N-major
  transpose_cvt<<<dim3(NG / 32, NINPUT / 32), dim3(32, 8), 0, stream>>>(W_ih, Wih_t, NINPUT, NG);
  transpose_cvt<<<dim3(NG / 32, NHID / 32),   dim3(32, 8), 0, stream>>>(W_hh, Whh_t, NHID, NG);
  {
    size_t n4 = (size_t)NVOC * NHID / 4;
    cvt_bf16<<<(unsigned)((n4 + 255) / 256), 256, 0, stream>>>(decW, decW_b, n4);
  }
  // xg = emb @ W_ih + b     (M=4096, N=4096, K=512)
  gemm_bf16_nt<<<dim3(NG / 128, MTOT / 128), 256, 0, stream>>>(
      emb_b, Wih_t, bias, xg, NINPUT, NINPUT, NINPUT, NG);
  // sequential recurrence: 256 steps, 2 kernels each
  for (int t = 0; t < TT; ++t) {
    lstm_gemm_hh<<<32, 256, 0, stream>>>(h_b, Whh_t, xg, gates, t);
    lstm_pointwise<<<64, 256, 0, stream>>>(gates, c_f, h_f, h_b, hall_b, t);
  }
  // decoder: out = h_all @ dec_W^T + dec_b   (M=4096, N=32000, K=1024)
  gemm_bf16_nt<<<dim3(NVOC / 128, MTOT / 128), 256, 0, stream>>>(
      hall_b, decW_b, decb, out, NHID, NHID, NHID, NVOC);
  // hidden state outputs appended after decoded
  copy_hidden<<<64, 256, 0, stream>>>(h_f, c_f, out + (size_t)MTOT * NVOC);
}
